// MultiHeadSelfAttention_15298673508593
// MI455X (gfx1250) — matrix operand order repaired and checked by reading, not compiled, not run on hardware
//
#include <hip/hip_runtime.h>
#include <hip/hip_bf16.h>

// Problem constants (from reference: B=2, T=2048, C=768, H=12, D=64)
#define BB 2
#define TT 2048
#define CC 768
#define HH 12
#define DD 64
#define C3 2304          // 3*C
#define BT (BB*TT)       // 4096 tokens
#define BH (BB*HH)       // 24 (batch*heads)

typedef __attribute__((ext_vector_type(8)))  float        v8f;
typedef __attribute__((ext_vector_type(16))) __bf16       v16bf;
typedef __attribute__((ext_vector_type(8)))  unsigned int v8u;
typedef __attribute__((ext_vector_type(4)))  unsigned int v4u;

// float -> bf16 (round to nearest even), raw 16-bit result in low bits
__device__ __forceinline__ unsigned int f2bf_raw(float f) {
    unsigned int u = __builtin_bit_cast(unsigned int, f);
    return (u + 0x7FFFu + ((u >> 16) & 1u)) >> 16;
}
__device__ __forceinline__ unsigned short f2bf(float f) {
    return (unsigned short)f2bf_raw(f);
}

__device__ __forceinline__ v8f wmma_bf16(v8u a, v8u b, v8f c) {
    return __builtin_amdgcn_wmma_f32_16x16x32_bf16(
        /*neg_a=*/false, __builtin_bit_cast(v16bf, a),
        /*neg_b=*/false, __builtin_bit_cast(v16bf, b),
        /*c_mod=*/(short)0, c, /*reuse_a=*/false, /*reuse_b=*/false);
}

// ---------------- packing kernels ----------------
__global__ void cvt_x_kernel(const float* __restrict__ x,
                             unsigned short* __restrict__ xb, int n) {
    int i = blockIdx.x * blockDim.x + threadIdx.x;
    if (i < n) xb[i] = f2bf(x[i]);
}

// wT[o*rows + c] = bf16(W[c*cols + o])   (W is rows x cols row-major)
__global__ void transpose_w_kernel(const float* __restrict__ W,
                                   unsigned short* __restrict__ wT,
                                   int rows, int cols) {
    int i = blockIdx.x * blockDim.x + threadIdx.x;
    if (i < rows * cols) {
        int o = i / rows;
        int c = i - o * rows;
        wT[(size_t)o * rows + c] = f2bf(W[(size_t)c * cols + o]);
    }
}

// ---------------- QKV GEMM: (BT x C) @ (C x 3C) + bias, scatter to Q/K/Vt ----
// One wave per 16x16 output tile. A = xb [token][c] bf16, B = WqkvT [out][c] bf16.
__global__ void qkv_gemm_kernel(const unsigned short* __restrict__ xb,
                                const unsigned short* __restrict__ wT,
                                const float* __restrict__ bqkv,
                                unsigned short* __restrict__ Qb,
                                unsigned short* __restrict__ Kb,
                                unsigned short* __restrict__ Vt) {
    const int lane = threadIdx.x & 31;
    const int wid  = (blockIdx.x * blockDim.x + threadIdx.x) >> 5;
    const int Ntiles = C3 / 16;                 // 144
    const int m0 = (wid / Ntiles) * 16;
    const int n0 = (wid - (wid / Ntiles) * Ntiles) * 16;
    const int ka = (lane >> 4) * 16;            // A K-chunk (elements)
    const int kb = (lane >> 4) * 16;            // B K-chunk (elements)
    const unsigned short* ap = xb + (size_t)(m0 + (lane & 15)) * CC;
    const unsigned short* bp = wT + (size_t)(n0 + (lane & 15)) * CC;

    v8f acc = {};
    for (int kk = 0; kk < CC; kk += 32) {
        if (kk + 32 < CC) {
            __builtin_prefetch(ap + kk + 32 + ka, 0, 1);
            __builtin_prefetch(bp + kk + 32 + kb, 0, 1);
        }
        v4u a_lo = *(const v4u*)(ap + kk + ka);
        v4u a_hi = *(const v4u*)(ap + kk + ka + 8);
        v8u av = {a_lo[0], a_lo[1], a_lo[2], a_lo[3],
                  a_hi[0], a_hi[1], a_hi[2], a_hi[3]};
        v8u bv = *(const v8u*)(bp + kk + kb);
        acc = wmma_bf16(av, bv, acc);
    }

    const int col  = n0 + (lane & 15);          // 0..2303
    const float bias = bqkv[col];
    const int sect = col / CC;                  // 0=q 1=k 2=v
    const int idx  = col - sect * CC;
    const int h = idx >> 6, d = idx & 63;
    const int mb = 8 * (lane >> 4);
#pragma unroll
    for (int r = 0; r < 8; ++r) {
        int m = m0 + mb + r;                    // token
        int b = m / TT, t = m - b * TT;
        size_t bh = (size_t)(b * HH + h);
        float v = acc[r] + bias;
        if (sect == 0)      Qb[(bh * TT + t) * DD + d] = f2bf(v * 0.125f); // 1/sqrt(64)
        else if (sect == 1) Kb[(bh * TT + t) * DD + d] = f2bf(v);
        else                Vt[(bh * DD + d) * TT + t] = f2bf(v);          // V transposed [d][t]
    }
}

// ---------------- scores: Q @ K^T, causal mask, fp32 into att ----------------
// Block = 8 waves owning 8 consecutive q-tiles of one (b,h); sweeps a 32-wide
// chunk of k-tiles. Each 16x64 K tile is async-copied into LDS once
// (GLOBAL_LOAD_ASYNC_TO_LDS_B128, ASYNCcnt), double-buffered so the copy of
// tile kt+1 overlaps the WMMAs on tile kt. Q fragments stay in registers.
// Note: reference's ALiBi uses clip(j-i,0) == 0 on all causally-kept entries,
// so the bias is a mathematical no-op and is omitted.
__global__ void scores_kernel(const unsigned short* __restrict__ Qb,
                              const unsigned short* __restrict__ Kb,
                              float* __restrict__ att) {
    __shared__ unsigned short sK[2][16 * DD];   // 2 x 2KB K tiles
    const int tid  = threadIdx.x;
    const int lane = tid & 31;
    const int w    = tid >> 5;                  // wave id 0..7

    const int bh      = blockIdx.x >> 6;        // / 64
    const int rem     = blockIdx.x & 63;
    const int qtGroup = rem >> 2;               // 0..15
    const int chunk   = rem & 3;                // 0..3 (32 k-tiles each)
    const int qt      = qtGroup * 8 + w;        // this wave's q-tile
    const int qtMax   = qtGroup * 8 + 7;
    const int c0      = chunk * 32;
    const int ktEnd   = min(c0 + 32, qtMax + 1);
    if (c0 >= ktEnd) return;                    // uniform over block

    // Q fragments for this wave's 16 query rows (K-dim = 64 => 2 fragments)
    const int q0 = qt * 16;
    const int ka = (lane >> 4) * 16;
    const unsigned short* qp = Qb + ((size_t)bh * TT + q0 + (lane & 15)) * DD;
    v8u qa[2];
#pragma unroll
    for (int i = 0; i < 2; ++i) {
        v4u lo = *(const v4u*)(qp + 32 * i + ka);
        v4u hi = *(const v4u*)(qp + 32 * i + ka + 8);
        qa[i] = (v8u){lo[0], lo[1], lo[2], lo[3], hi[0], hi[1], hi[2], hi[3]};
    }

    const bool loader = (tid < 128);            // waves 0..3: 128 x 16B = 2KB tile
    const unsigned short* ksrc = Kb + (size_t)bh * TT * DD + (size_t)tid * 8;

    // issue async copy of K tile kt into buffer (kt&1)
    auto issue = [&](int kt) {
        if (loader) {
            const unsigned short* src = ksrc + (size_t)kt * 16 * DD;
            unsigned dst = (unsigned)(size_t)&sK[kt & 1][tid * 8];
            asm volatile("global_load_async_to_lds_b128 %0, %1, off"
                         :: "v"(dst), "v"(src) : "memory");
        }
    };

    issue(c0);
    float* ap = att + (size_t)bh * TT * TT;
    for (int kt = c0; kt < ktEnd; ++kt) {
        if (loader) asm volatile("s_wait_asynccnt 0x0" ::: "memory");
        __syncthreads();                        // tile kt visible to all waves
        if (kt + 1 < ktEnd) issue(kt + 1);      // overlap next copy with compute

        if (kt <= qt) {
            const unsigned short* kb0 =
                &sK[kt & 1][(lane & 15) * DD + (lane >> 4) * 16];
            v8f acc = {};
#pragma unroll
            for (int i = 0; i < 2; ++i) {
                v8u bv = *(const v8u*)(kb0 + 32 * i);
                acc = wmma_bf16(qa[i], bv, acc);
            }
            const int j = kt * 16 + (lane & 15);
#pragma unroll
            for (int r = 0; r < 8; ++r) {
                int i2 = q0 + 8 * (lane >> 4) + r;
                ap[(size_t)i2 * TT + j] = (j > i2) ? -__builtin_inff() : acc[r];
            }
        }
        __syncthreads();                        // everyone done with buffer (kt&1)
    }
}

// ---------------- row softmax, whole row staged in LDS ----------------------
__global__ void softmax_kernel(float* __restrict__ att) {
    __shared__ float srow[TT];
    __shared__ float sred[256];
    const int row = blockIdx.x;                 // bh*T + i
    const int i   = row & (TT - 1);
    const int tid = threadIdx.x;
    float* ap = att + (size_t)row * TT;
    const int L = i + 1;

    float mx = -__builtin_inff();
    for (int j = tid; j < L; j += 256) { float v = ap[j]; srow[j] = v; mx = fmaxf(mx, v); }
    sred[tid] = mx; __syncthreads();
    for (int s = 128; s > 0; s >>= 1) {
        if (tid < s) sred[tid] = fmaxf(sred[tid], sred[tid + s]);
        __syncthreads();
    }
    mx = sred[0]; __syncthreads();

    float sum = 0.f;
    for (int j = tid; j < L; j += 256) { float e = __expf(srow[j] - mx); srow[j] = e; sum += e; }
    sred[tid] = sum; __syncthreads();
    for (int s = 128; s > 0; s >>= 1) {
        if (tid < s) sred[tid] += sred[tid + s];
        __syncthreads();
    }
    const float inv = 1.0f / sred[0];
    for (int j = tid; j < L; j += 256)      ap[j] = srow[j] * inv;
    for (int j = L + tid; j < TT; j += 256) ap[j] = 0.f;     // masked entries
}

// ---------------- att @ V -> y_heads bf16 [token][h*D+d] -------------------
__global__ void attv_kernel(const float* __restrict__ att,
                            const unsigned short* __restrict__ Vt,
                            unsigned short* __restrict__ yh) {
    const int lane = threadIdx.x & 31;
    const int wid  = (blockIdx.x * blockDim.x + threadIdx.x) >> 5;
    const int dt = wid & 3;
    const int qt = (wid >> 2) & 127;
    const int bh = wid >> 9;
    const int q0 = qt * 16, d0 = dt * 16;
    const float*          ap0 = att + ((size_t)bh * TT + q0 + (lane & 15)) * TT + (lane >> 4) * 16;
    const unsigned short* bp0 = Vt  + ((size_t)bh * DD + d0 + (lane & 15)) * TT + (lane >> 4) * 16;

    v8f acc = {};
    const int limit = q0 + 16;                  // keys beyond row are exactly 0
    for (int kk = 0; kk < limit; kk += 32) {
        v8f f0 = *(const v8f*)(ap0 + kk);
        v8f f1 = *(const v8f*)(ap0 + kk + 8);
        v8u av;
#pragma unroll
        for (int p = 0; p < 4; ++p) {
            av[p]     = f2bf_raw(f0[2 * p]) | (f2bf_raw(f0[2 * p + 1]) << 16);
            av[4 + p] = f2bf_raw(f1[2 * p]) | (f2bf_raw(f1[2 * p + 1]) << 16);
        }
        v8u bv = *(const v8u*)(bp0 + kk);
        acc = wmma_bf16(av, bv, acc);
    }

    const int b = bh / HH, h = bh - b * HH;
    const int dcol = d0 + (lane & 15);
#pragma unroll
    for (int r = 0; r < 8; ++r) {
        int q = q0 + 8 * (lane >> 4) + r;
        yh[((size_t)b * TT + q) * CC + h * DD + dcol] = f2bf(acc[r]);
    }
}

// ---------------- projection: yh @ Wproj + bias -> fp32 y -------------------
__global__ void proj_gemm_kernel(const unsigned short* __restrict__ yh,
                                 const unsigned short* __restrict__ wT,
                                 const float* __restrict__ bproj,
                                 float* __restrict__ y) {
    const int lane = threadIdx.x & 31;
    const int wid  = (blockIdx.x * blockDim.x + threadIdx.x) >> 5;
    const int Ntiles = CC / 16;                 // 48
    const int m0 = (wid / Ntiles) * 16;
    const int n0 = (wid - (wid / Ntiles) * Ntiles) * 16;
    const int ka = (lane >> 4) * 16;
    const int kb = (lane >> 4) * 16;
    const unsigned short* ap = yh + (size_t)(m0 + (lane & 15)) * CC;
    const unsigned short* bp = wT + (size_t)(n0 + (lane & 15)) * CC;

    v8f acc = {};
    for (int kk = 0; kk < CC; kk += 32) {
        if (kk + 32 < CC) {
            __builtin_prefetch(ap + kk + 32 + ka, 0, 1);
            __builtin_prefetch(bp + kk + 32 + kb, 0, 1);
        }
        v4u a_lo = *(const v4u*)(ap + kk + ka);
        v4u a_hi = *(const v4u*)(ap + kk + ka + 8);
        v8u av = {a_lo[0], a_lo[1], a_lo[2], a_lo[3],
                  a_hi[0], a_hi[1], a_hi[2], a_hi[3]};
        v8u bv = *(const v8u*)(bp + kk + kb);
        acc = wmma_bf16(av, bv, acc);
    }

    const int col = n0 + (lane & 15);
    const float bias = bproj[col];
#pragma unroll
    for (int r = 0; r < 8; ++r) {
        int m = m0 + 8 * (lane >> 4) + r;
        y[(size_t)m * CC + col] = acc[r] + bias;
    }
}

extern "C" void kernel_launch(void* const* d_in, const int* in_sizes, int n_in,
                              void* d_out, int out_size, void* d_ws, size_t ws_size,
                              hipStream_t stream) {
    (void)in_sizes; (void)n_in; (void)out_size; (void)ws_size;
    const float* x     = (const float*)d_in[0];
    const float* Wqkv  = (const float*)d_in[1];
    const float* bqkv  = (const float*)d_in[2];
    const float* Wproj = (const float*)d_in[3];
    const float* bproj = (const float*)d_in[4];

    float* y   = (float*)d_out;                 // (B,T,C) fp32
    float* att = y + (size_t)BT * CC;           // (B,H,T,T) fp32

    char* ws = (char*)d_ws;
    size_t off = 0;
    auto salloc = [&](size_t bytes) {
        void* p = ws + off;
        off += (bytes + 255) & ~(size_t)255;
        return p;
    };
    unsigned short* xb     = (unsigned short*)salloc((size_t)BT * CC * 2);
    unsigned short* wqkvT  = (unsigned short*)salloc((size_t)C3 * CC * 2);
    unsigned short* wprojT = (unsigned short*)salloc((size_t)CC * CC * 2);
    unsigned short* Qb     = (unsigned short*)salloc((size_t)BH * TT * DD * 2);
    unsigned short* Kb     = (unsigned short*)salloc((size_t)BH * TT * DD * 2);
    unsigned short* Vt     = (unsigned short*)salloc((size_t)BH * DD * TT * 2);
    unsigned short* yh     = (unsigned short*)salloc((size_t)BT * CC * 2);

    cvt_x_kernel<<<(BT * CC + 255) / 256, 256, 0, stream>>>(x, xb, BT * CC);
    transpose_w_kernel<<<(CC * C3 + 255) / 256, 256, 0, stream>>>(Wqkv, wqkvT, CC, C3);
    transpose_w_kernel<<<(CC * CC + 255) / 256, 256, 0, stream>>>(Wproj, wprojT, CC, CC);

    qkv_gemm_kernel<<<(BT / 16) * (C3 / 16) / 8, 256, 0, stream>>>(xb, wqkvT, bqkv, Qb, Kb, Vt);
    scores_kernel<<<BH * 16 * 4, 256, 0, stream>>>(Qb, Kb, att);
    softmax_kernel<<<BH * TT, 256, 0, stream>>>(att);
    attv_kernel<<<BH * 128 * 4 / 8, 256, 0, stream>>>(att, Vt, yh);
    proj_gemm_kernel<<<(BT / 16) * (CC / 16) / 8, 256, 0, stream>>>(yh, wprojT, bproj, y);
}
